// ReAttention_6803228196954
// MI455X (gfx1250) — compile-verified, matrix operand order checked by reading, not run on hardware
//
#include <hip/hip_runtime.h>
#include <hip/hip_bf16.h>
#include <hip/hip_fp16.h>

// ---------------- problem constants ----------------
#define BB 8
#define NN 1024
#define CC 768
#define HH 8
#define DD 96          // head dim
#define SS 16
#define QK_SCALE 0.1020620726159658f   // 96^-0.5

typedef __attribute__((ext_vector_type(16))) _Float16 v16h;
typedef __attribute__((ext_vector_type(8)))  float    v8f;

// ---------------- WMMA helpers (CDNA5 wave32) ----------------
// 16-bit A-fragment (16x32 MxK) per ISA layout:
//   lane l: row M = l&15 ; element i -> K = i + (i&8) + 8*(l>>4)
// B-fragment (32x16 KxN) is the transpose-symmetric layout, so loading a
// row-major [N][K] matrix with the same pattern yields B for D = A x B^T-style
// GEMMs (B stored as rows of the "other" operand).
__device__ __forceinline__ v16h load_frag_f16(const _Float16* __restrict__ p,
                                              int ld, int lane) {
  const _Float16* r = p + (size_t)(lane & 15) * ld + ((lane >> 4) << 3);
  v16h f;
#pragma unroll
  for (int i = 0; i < 16; ++i) f[i] = r[i + (i & 8)];
  return f;
}

__device__ __forceinline__ v16h load_frag_f32_cvt(const float* __restrict__ p,
                                                  int ld, int lane) {
  const float* r = p + (size_t)(lane & 15) * ld + ((lane >> 4) << 3);
  v16h f;
#pragma unroll
  for (int i = 0; i < 16; ++i) f[i] = (_Float16)r[i + (i & 8)];
  return f;
}

__device__ __forceinline__ v8f wmma16(v16h a, v16h b, v8f c) {
  // v_wmma_f32_16x16x32_f16 : D = A(16x32) * B(32x16) + C
  return __builtin_amdgcn_wmma_f32_16x16x32_f16(
      false, a, false, b, (short)0, c, false, false);
}

// C/D layout: lane l, reg r -> M = r + 8*(l>>4), N = l&15

// ---------------- K1: 3x3 convs -> q,k,v (f16, attention-friendly layouts) --
__global__ __launch_bounds__(256) void k_qkv_conv(
    const float* __restrict__ x,
    const float* __restrict__ qw, const float* __restrict__ qb,
    const float* __restrict__ kw, const float* __restrict__ kb,
    const float* __restrict__ vw, const float* __restrict__ vb,
    _Float16* __restrict__ qh,   // (B,H,N,D) pre-scaled by QK_SCALE
    _Float16* __restrict__ kh,   // (B,H,N,D)
    _Float16* __restrict__ vT) { // (B,H,D,N)
  __shared__ float xin[3 * SS * SS];
  __shared__ float w[3][81];     // [conv][o*27 + i*9 + ky*3 + kx]
  __shared__ float bias[3][3];

  const int tid = threadIdx.x;
  const int tok = blockIdx.x;          // b*N + n
  const int b = tok >> 10, n = tok & (NN - 1);

  const float* xp = x + (size_t)tok * CC;
#pragma unroll
  for (int i = tid; i < CC; i += 256) xin[i] = xp[i];
  if (tid < 81) {
    w[0][tid] = qw[tid]; w[1][tid] = kw[tid]; w[2][tid] = vw[tid];
  }
  if (tid < 3) { bias[0][tid] = qb[tid]; bias[1][tid] = kb[tid]; bias[2][tid] = vb[tid]; }
  __syncthreads();

  const int s0 = tid >> 4, s1 = tid & 15;
  float res[3][3];
#pragma unroll
  for (int cv = 0; cv < 3; ++cv) {
#pragma unroll
    for (int o = 0; o < 3; ++o) {
      float acc = bias[cv][o];
#pragma unroll
      for (int ic = 0; ic < 3; ++ic) {
#pragma unroll
        for (int ky = 0; ky < 3; ++ky) {
#pragma unroll
          for (int kx = 0; kx < 3; ++kx) {
            int yy = s0 + ky - 1, xx = s1 + kx - 1;
            if ((unsigned)yy < SS && (unsigned)xx < SS)
              acc += w[cv][o * 27 + ic * 9 + ky * 3 + kx] *
                     xin[ic * 256 + yy * SS + xx];
          }
        }
      }
      res[cv][o] = acc;
    }
  }

#pragma unroll
  for (int o = 0; o < 3; ++o) {
    const int c = o * 256 + tid;         // channel in [0,768)
    const int h = c / DD, d = c - h * DD;
    const size_t bh = (size_t)(b * HH + h);
    qh[(bh * NN + n) * DD + d] = (_Float16)(res[0][o] * QK_SCALE);
    kh[(bh * NN + n) * DD + d] = (_Float16)res[1][o];
    vT[(bh * DD + d) * NN + n] = (_Float16)res[2][o];
  }
}

// ---------------- K2: scores = q k^T, softmax numerator + row sums ---------
__global__ __launch_bounds__(256) void k_attn_scores(
    const _Float16* __restrict__ qh, const _Float16* __restrict__ kh,
    float* __restrict__ P,            // (B,H,N,N) unnormalized exp (attn region)
    float* __restrict__ rsum) {       // (B,H,N)
  const int lane = threadIdx.x & 31, wv = threadIdx.x >> 5;
  const int h = blockIdx.y, b = blockIdx.z;
  const size_t bh = (size_t)(b * HH + h);
  const int n0 = blockIdx.x * 128 + wv * 16;

  const _Float16* qbase = qh + (bh * NN + n0) * DD;
  const _Float16* kbase = kh + bh * NN * DD;

  v16h aq[3];
#pragma unroll
  for (int j = 0; j < 3; ++j) aq[j] = load_frag_f16(qbase + j * 32, DD, lane);

  float rmax[8];
#pragma unroll
  for (int r = 0; r < 8; ++r) rmax[r] = -3.0e38f;

  // pass 1: row maxima
  for (int m0 = 0; m0 < NN; m0 += 16) {
    const _Float16* kc = kbase + (size_t)m0 * DD;
    __builtin_prefetch(kc + 16 * DD, 0, 1);
    v8f acc = {};
#pragma unroll
    for (int j = 0; j < 3; ++j)
      acc = wmma16(aq[j], load_frag_f16(kc + j * 32, DD, lane), acc);
#pragma unroll
    for (int r = 0; r < 8; ++r) rmax[r] = fmaxf(rmax[r], acc[r]);
  }
#pragma unroll
  for (int m = 1; m <= 8; m <<= 1)
#pragma unroll
    for (int r = 0; r < 8; ++r)
      rmax[r] = fmaxf(rmax[r], __shfl_xor(rmax[r], m, 32));

  // pass 2: exp(s - max), accumulate sums, stream numerators to HBM
  float rs[8];
#pragma unroll
  for (int r = 0; r < 8; ++r) rs[r] = 0.f;
  float* Pbase = P + (bh * NN + n0) * (size_t)NN;
  const int row_hi = (lane >> 4) << 3, col0 = lane & 15;

  for (int m0 = 0; m0 < NN; m0 += 16) {
    const _Float16* kc = kbase + (size_t)m0 * DD;
    __builtin_prefetch(kc + 16 * DD, 0, 1);
    v8f acc = {};
#pragma unroll
    for (int j = 0; j < 3; ++j)
      acc = wmma16(aq[j], load_frag_f16(kc + j * 32, DD, lane), acc);
#pragma unroll
    for (int r = 0; r < 8; ++r) {
      float e = __expf(acc[r] - rmax[r]);
      rs[r] += e;
      Pbase[(size_t)(r + row_hi) * NN + (m0 + col0)] = e;
    }
  }
#pragma unroll
  for (int m = 1; m <= 8; m <<= 1)
#pragma unroll
    for (int r = 0; r < 8; ++r) rs[r] += __shfl_xor(rs[r], m, 32);
  if ((lane & 15) == 0) {
#pragma unroll
    for (int r = 0; r < 8; ++r)
      rsum[bh * NN + n0 + r + row_hi] = rs[r];
  }
}

// ---------------- K3: normalize + head mix (rw) + eval BatchNorm, in place --
__global__ __launch_bounds__(256) void k_reattn(
    float* __restrict__ attn, const float* __restrict__ rsum,
    const float* __restrict__ rw, const float* __restrict__ rb,
    const float* __restrict__ bn_g, const float* __restrict__ bn_b,
    const float* __restrict__ bn_m, const float* __restrict__ bn_v) {
  __shared__ float srw[64], sA[8], sC[8];
  const int tid = threadIdx.x;
  if (tid < 64) srw[tid] = rw[tid];
  if (tid < 8) {
    float inv = bn_g[tid] * rsqrtf(bn_v[tid] + 1e-5f);
    sA[tid] = inv;
    sC[tid] = (rb[tid] - bn_m[tid]) * inv + bn_b[tid];
  }
  __syncthreads();

  const size_t idx = (size_t)blockIdx.x * 256 + tid;   // b*N*N + n*N + m
  const int b = (int)(idx >> 20);
  const size_t rem = idx & ((1u << 20) - 1);
  const int n = (int)(rem >> 10);
  const size_t base = ((size_t)(b * HH)) << 20;        // b*H*N*N
  const size_t off = rem;                              // n*N + m

  float p[8];
#pragma unroll
  for (int h = 0; h < 8; ++h) {
    float inv = 1.0f / rsum[((size_t)(b * HH + h)) * NN + n];
    p[h] = attn[base + ((size_t)h << 20) + off] * inv;
  }
#pragma unroll
  for (int o = 0; o < 8; ++o) {
    float acc = 0.f;
#pragma unroll
    for (int h = 0; h < 8; ++h) acc += srw[o * 8 + h] * p[h];
    attn[base + ((size_t)o << 20) + off] = acc * sA[o] + sC[o];
  }
}

// ---------------- K4: out_pre(B,N,C) = attn_mixed @ v ----------------------
__global__ __launch_bounds__(256) void k_attn_v(
    const float* __restrict__ attn, const _Float16* __restrict__ vT,
    float* __restrict__ outpre) {
  const int lane = threadIdx.x & 31, wv = threadIdx.x >> 5;
  const int h = blockIdx.y, b = blockIdx.z;
  const size_t bh = (size_t)(b * HH + h);
  const int n0 = blockIdx.x * 128 + wv * 16;

  const float* Abase = attn + (bh * NN + n0) * (size_t)NN;  // rows n, cols m
  const _Float16* Vbase = vT + bh * DD * NN;                // rows d, cols m

  v8f acc[6] = {};
  for (int m0 = 0; m0 < NN; m0 += 32) {
    __builtin_prefetch(Abase + m0 + 32, 0, 1);
    v16h a = load_frag_f32_cvt(Abase + m0, NN, lane);
#pragma unroll
    for (int dt = 0; dt < 6; ++dt) {
      v16h bv = load_frag_f16(Vbase + (size_t)(dt * 16) * NN + m0, NN, lane);
      acc[dt] = wmma16(a, bv, acc[dt]);
    }
  }
  const int row_hi = (lane >> 4) << 3, col = lane & 15;
#pragma unroll
  for (int dt = 0; dt < 6; ++dt)
#pragma unroll
    for (int r = 0; r < 8; ++r) {
      int nrow = n0 + r + row_hi;
      int d = dt * 16 + col;
      outpre[((size_t)(b * NN + nrow)) * CC + h * DD + d] = acc[dt][r];
    }
}

// ---------------- K5: out = out_pre @ pw^T + pb ----------------------------
__global__ __launch_bounds__(256) void k_proj(
    const float* __restrict__ outpre, const float* __restrict__ pw,
    const float* __restrict__ pb, float* __restrict__ out) {
  const int lane = threadIdx.x & 31, wv = threadIdx.x >> 5;
  const int r0 = blockIdx.x * 128 + wv * 16;   // token rows
  const int c0 = blockIdx.y * 128;             // output channels

  const float* Abase = outpre + (size_t)r0 * CC;
  v8f acc[8] = {};
  for (int k0 = 0; k0 < CC; k0 += 32) {
    __builtin_prefetch(Abase + k0 + 32, 0, 1);
    v16h a = load_frag_f32_cvt(Abase + k0, CC, lane);
#pragma unroll
    for (int ct = 0; ct < 8; ++ct) {
      // B[K=cin][N=cout] = pw[cout][cin]; pw rows are contiguous in cin
      v16h bw = load_frag_f32_cvt(pw + (size_t)(c0 + ct * 16) * CC + k0, CC, lane);
      acc[ct] = wmma16(a, bw, acc[ct]);
    }
  }
  const int row_hi = (lane >> 4) << 3, col = lane & 15;
#pragma unroll
  for (int ct = 0; ct < 8; ++ct)
#pragma unroll
    for (int r = 0; r < 8; ++r) {
      int rr = r0 + r + row_hi;
      int cc = c0 + ct * 16 + col;
      out[(size_t)rr * CC + cc] = acc[ct][r] + pb[cc];
    }
}

// ---------------- host launcher --------------------------------------------
extern "C" void kernel_launch(void* const* d_in, const int* in_sizes, int n_in,
                              void* d_out, int out_size, void* d_ws, size_t ws_size,
                              hipStream_t stream) {
  (void)in_sizes; (void)n_in; (void)out_size; (void)ws_size;
  const float* x   = (const float*)d_in[0];
  const float* qw  = (const float*)d_in[1];
  const float* qb  = (const float*)d_in[2];
  const float* kw  = (const float*)d_in[3];
  const float* kb  = (const float*)d_in[4];
  const float* vw  = (const float*)d_in[5];
  const float* vb  = (const float*)d_in[6];
  const float* rw  = (const float*)d_in[7];
  const float* rb  = (const float*)d_in[8];
  const float* bng = (const float*)d_in[9];
  const float* bnb = (const float*)d_in[10];
  const float* bnm = (const float*)d_in[11];
  const float* bnv = (const float*)d_in[12];
  const float* pw  = (const float*)d_in[13];
  const float* pb  = (const float*)d_in[14];

  float* out  = (float*)d_out;                       // (B,N,C)
  float* attn = out + (size_t)BB * NN * CC;          // (B,H,N,N)

  // workspace layout
  const size_t qkvN = (size_t)BB * HH * NN * DD;     // 6,291,456
  _Float16* qh = (_Float16*)d_ws;
  _Float16* kh = qh + qkvN;
  _Float16* vT = kh + qkvN;
  float* rsum   = (float*)(vT + qkvN);
  float* outpre = rsum + (size_t)BB * HH * NN;

  k_qkv_conv<<<BB * NN, 256, 0, stream>>>(x, qw, qb, kw, kb, vw, vb, qh, kh, vT);
  k_attn_scores<<<dim3(NN / 128, HH, BB), 256, 0, stream>>>(qh, kh, attn, rsum);
  k_reattn<<<(BB * NN * NN) / 256, 256, 0, stream>>>(attn, rsum, rw, rb,
                                                     bng, bnb, bnm, bnv);
  k_attn_v<<<dim3(NN / 128, HH, BB), 256, 0, stream>>>(attn, vT, outpre);
  k_proj<<<dim3((BB * NN) / 128, CC / 128), 256, 0, stream>>>(outpre, pw, pb, out);
}
